// Transformer_936302870566
// MI455X (gfx1250) — matrix operand order repaired and checked by reading, not compiled, not run on hardware
//
#include <hip/hip_runtime.h>
#include <math.h>

typedef _Float16 half_t;
typedef __attribute__((ext_vector_type(16))) _Float16 v16h;
typedef __attribute__((ext_vector_type(8)))  _Float16 v8h;
typedef __attribute__((ext_vector_type(8)))  float    v8f;
typedef __attribute__((ext_vector_type(4)))  int      v4i;

#define WMMA_F16(a, b, c) \
  __builtin_amdgcn_wmma_f32_16x16x32_f16(false, (a), false, (b), (short)0, (c), false, false)

#ifndef __has_builtin
#define __has_builtin(x) 0
#endif
#if __has_builtin(__builtin_amdgcn_global_load_async_to_lds_b128)
#define ASYNC_LDS 1
#else
#define ASYNC_LDS 0
#endif

// Problem constants (match reference)
constexpr int Bn  = 4;
constexpr int Mn  = 2048;
constexpr int Cn  = 1024;
constexpr int Hn  = 8;
constexpr int DKn = 32;    // per-head q/k dim
constexpr int DVn = 128;   // per-head v dim
constexpr int NTOT = 1536; // 256 (Q) + 256 (K) + 1024 (V) projection columns

// ---------------------------------------------------------------------------
// Async 16-byte global->LDS copy (CDNA5 GLOBAL_LOAD_ASYNC_TO_LDS_B128,
// tracked by ASYNCcnt). Builtin signature takes v4i addrspace(1)* source and
// v4i addrspace(3)* LDS destination. Generic LDS pointers carry the LDS byte
// offset in their low 32 bits (ISA 10.2 aperture mapping), so truncation
// yields the as(3) address. Fallback: ordinary load + ds_store.
// ---------------------------------------------------------------------------
__device__ inline void async_cp16(const half_t* gsrc, half_t* ldst) {
#if ASYNC_LDS
  __builtin_amdgcn_global_load_async_to_lds_b128(
      (__attribute__((address_space(1))) v4i*)(uintptr_t)gsrc,
      (__attribute__((address_space(3))) v4i*)(unsigned)(uintptr_t)ldst,
      0, 0);
#else
  *(uint4*)ldst = *(const uint4*)gsrc;
#endif
}

__device__ inline void wait_async_lds() {
#if ASYNC_LDS
  asm volatile("s_wait_asynccnt 0x0" ::: "memory");
#endif
}

// ---------------------------------------------------------------------------
// Fragment loaders, following CDNA5 16-bit WMMA VGPR layouts (ISA 7.12.2):
//  A (16x32, MxK): lanes 0-15 row=lane, elems = K{klo..klo+7, 16+klo..23+klo},
//                  klo = 0 for lanes<16, 8 for lanes>=16.
//  B (32x16, KxN): lane = col (mod 16), elems = K{klo..klo+7, 16+klo..23+klo},
//                  klo = 0 for lanes<16, 8 for lanes>=16.
// ---------------------------------------------------------------------------
__device__ inline v16h load_a_f32(const float* base, int ld, int k0) {
  unsigned lane = threadIdx.x & 31u;
  unsigned row  = lane & 15u;
  unsigned klo  = (lane < 16u) ? 0u : 8u;
  const float* p = base + (size_t)row * ld + k0 + klo;
  float4 a0 = *(const float4*)(p);
  float4 a1 = *(const float4*)(p + 4);
  float4 b0 = *(const float4*)(p + 16);
  float4 b1 = *(const float4*)(p + 20);
  v16h r;
  r[0]=(half_t)a0.x; r[1]=(half_t)a0.y; r[2]=(half_t)a0.z; r[3]=(half_t)a0.w;
  r[4]=(half_t)a1.x; r[5]=(half_t)a1.y; r[6]=(half_t)a1.z; r[7]=(half_t)a1.w;
  r[8]=(half_t)b0.x; r[9]=(half_t)b0.y; r[10]=(half_t)b0.z; r[11]=(half_t)b0.w;
  r[12]=(half_t)b1.x; r[13]=(half_t)b1.y; r[14]=(half_t)b1.z; r[15]=(half_t)b1.w;
  return r;
}

__device__ inline v16h load_a_h(const half_t* base, int ld, int k0) {
  unsigned lane = threadIdx.x & 31u;
  unsigned row  = lane & 15u;
  unsigned klo  = (lane < 16u) ? 0u : 8u;
  const half_t* p = base + (size_t)row * ld + k0 + klo;
  v8h lo = *(const v8h*)(p);
  v8h hi = *(const v8h*)(p + 16);
  v16h r;
#pragma unroll
  for (int i = 0; i < 8; ++i) { r[i] = lo[i]; r[8 + i] = hi[i]; }
  return r;
}

// B fragment from an LDS tile stored as [col][k] with 32 k per row.
__device__ inline v16h lds_b_frag(const half_t* tile, int col16) {
  unsigned lane = threadIdx.x & 31u;
  unsigned col  = lane & 15u;
  const half_t* p = tile + (size_t)(col16 * 16 + col) * 32 + ((lane < 16u) ? 0 : 8);
  v8h lo = *(const v8h*)(p);
  v8h hi = *(const v8h*)(p + 16);
  v16h r;
#pragma unroll
  for (int i = 0; i < 8; ++i) { r[i] = lo[i]; r[8 + i] = hi[i]; }
  return r;
}

// ---------------------------------------------------------------------------
// Kernel 1: pack [Wq|Wk|Wv] -> Wt[n][k] (f16, N-major) + fused bias array.
// ---------------------------------------------------------------------------
__global__ void __launch_bounds__(256)
pack_weights(const float* __restrict__ Wq, const float* __restrict__ Wk,
             const float* __restrict__ Wv, const float* __restrict__ bq,
             const float* __restrict__ bk, const float* __restrict__ bv,
             half_t* __restrict__ Wt, float* __restrict__ biasAll) {
  int idx = blockIdx.x * 256 + threadIdx.x;
  int total = NTOT * Cn;
  if (idx < total) {
    int n = idx / Cn;
    int k = idx - n * Cn;
    float w = (n < 256) ? Wq[k * 256 + n]
            : (n < 512) ? Wk[k * 256 + (n - 256)]
                        : Wv[k * 1024 + (n - 512)];
    Wt[idx] = (half_t)w;
  }
  if (idx < NTOT) {
    biasAll[idx] = (idx < 256) ? bq[idx]
                 : (idx < 512) ? bk[idx - 256]
                               : bv[idx - 512];
  }
}

// ---------------------------------------------------------------------------
// Kernel 2: QKV projection GEMM. Block = 128 rows x 64 cols (8 waves x 16
// rows). The 64x32 weight tile per K-step is staged in LDS with async copies
// (double buffered; copy of tile i+1 overlaps WMMA on tile i).
// ---------------------------------------------------------------------------
__global__ void __launch_bounds__(256)
qkv_gemm(const float* __restrict__ x, const half_t* __restrict__ Wt,
         const float* __restrict__ biasAll, half_t* __restrict__ Q,
         half_t* __restrict__ K, half_t* __restrict__ Vt) {
  __shared__ half_t Wb[2][64 * 32];            // 2 x 4 KB

  int tid  = threadIdx.x;
  int wave = tid >> 5;
  unsigned lane = tid & 31u;
  unsigned colL = lane & 15u;
  unsigned rAdd = (lane < 16u) ? 0u : 8u;

  int blk = blockIdx.x;                        // 64 row-blocks * 24 col groups
  int rb  = blk & 63;
  int ng  = blk >> 6;
  int r0  = rb * 128 + wave * 16;              // this wave's 16 rows
  int n0g = ng * 64;

  const float* xrow = x + (size_t)r0 * Cn;

  // Stage one 64(n) x 32(k) tile: thread t copies 16B chunk (n = t/4, c = t%4).
  auto stage = [&](int k0, int buf) {
    int n = tid >> 2;
    int c = tid & 3;
    async_cp16(Wt + (size_t)(n0g + n) * Cn + k0 + c * 8,
               &Wb[buf][n * 32 + c * 8]);
  };

  v8f acc[4];
#pragma unroll
  for (int t = 0; t < 4; ++t)
#pragma unroll
    for (int j = 0; j < 8; ++j) acc[t][j] = 0.f;

  stage(0, 0);
  wait_async_lds();
  __syncthreads();

  for (int ks = 0; ks < 32; ++ks) {            // K = 1024 in steps of 32
    int buf = ks & 1;
    if (ks + 1 < 32) stage((ks + 1) * 32, buf ^ 1);

    v16h a = load_a_f32(xrow, Cn, ks * 32);
#pragma unroll
    for (int t = 0; t < 4; ++t) {
      v16h b = lds_b_frag(&Wb[buf][0], t);
      acc[t] = WMMA_F16(a, b, acc[t]);
    }

    wait_async_lds();                          // next tile resident
    __syncthreads();                           // everyone done with buf
  }

#pragma unroll
  for (int t = 0; t < 4; ++t) {
    int gn = n0g + t * 16 + (int)colL;
    float bias = biasAll[gn];
#pragma unroll
    for (int j = 0; j < 8; ++j) {
      int grow = r0 + j + (int)rAdd;           // global row over B*M
      int bb = grow >> 11;
      int m  = grow & 2047;
      half_t hv = (half_t)(acc[t][j] + bias);
      if (gn < 256) {
        int h = gn >> 5, dk = gn & 31;
        Q[(((size_t)bb * Hn + h) * Mn + m) * DKn + dk] = hv;
      } else if (gn < 512) {
        int n2 = gn - 256;
        int h = n2 >> 5, dk = n2 & 31;
        K[(((size_t)bb * Hn + h) * Mn + m) * DKn + dk] = hv;
      } else {
        int n2 = gn - 512;
        int h = n2 >> 7, dv = n2 & 127;
        Vt[(((size_t)bb * Hn + h) * DVn + dv) * Mn + m] = hv;
      }
    }
  }
}

// ---------------------------------------------------------------------------
// Kernel 3: attention. Block = one (b,h) x 128 queries; 8 waves each own a
// 16-query tile but SHARE the 32-key K/V tiles, which are staged in LDS by
// async copies (double buffered). Per 32-key step: 2 QK^T WMMA -> p =
// mask*exp(relu(s/sqrt(DK))) -> LDS layout shuffle -> 8 PV WMMA (DV=128).
// Softmax denominator cancels against the post-mask renormalization.
// ---------------------------------------------------------------------------
__global__ void __launch_bounds__(256)
attn(const half_t* __restrict__ Q, const half_t* __restrict__ K,
     const half_t* __restrict__ Vt, const int* __restrict__ mask,
     const float* __restrict__ x, const float* __restrict__ gamma,
     float* __restrict__ out) {
  __shared__ half_t Kbuf[2][32 * 32];          // 2 x 2 KB, [key][dk]
  __shared__ half_t Vbuf[2][128 * 32];         // 2 x 8 KB, [dv][key]
  __shared__ half_t Pb[8][16 * 32];            // per-wave P tile, 8 KB

  int tid  = threadIdx.x;
  int wave = tid >> 5;
  unsigned lane = tid & 31u;
  unsigned colL = lane & 15u;
  unsigned rAdd = (lane < 16u) ? 0u : 8u;

  int blk = blockIdx.x;                        // 16 q-blocks * 32 (b,h)
  int qb  = blk & 15;
  int bh  = blk >> 4;
  int b = bh >> 3;
  int h = bh & 7;
  int qt = qb * 8 + wave;                      // this wave's 16-query tile

  const half_t* Qh = Q + ((size_t)b * Hn + h) * Mn * DKn;
  const half_t* Kh = K + ((size_t)b * Hn + h) * Mn * DKn;
  const half_t* Vh = Vt + ((size_t)b * Hn + h) * DVn * Mn;
  const int* mrow = mask + (size_t)b * Mn;
  const float scale = 0.1767766952966369f;     // 1/sqrt(32)

  // Stage a 32-key tile of K (2 KB contiguous) and V^T (128 dv x 64B rows).
  auto stage = [&](int kt32, int buf) {
    int k0 = kt32 * 32;
    if (tid < 128)                             // K: 128 x 16B chunks
      async_cp16(Kh + (size_t)k0 * DKn + tid * 8, &Kbuf[buf][tid * 8]);
    int dv = tid >> 1;                         // V: 2 x 16B per thread
    int hf = tid & 1;
    const half_t* vsrc = Vh + (size_t)dv * Mn + k0 + hf * 16;
    half_t* vdst = &Vbuf[buf][dv * 32 + hf * 16];
    async_cp16(vsrc, vdst);
    async_cp16(vsrc + 8, vdst + 8);
  };

  v16h aq = load_a_h(Qh + (size_t)qt * 16 * DKn, DKn, 0);

  v8f acc[8];
  float rsum[8];
#pragma unroll
  for (int jt = 0; jt < 8; ++jt)
#pragma unroll
    for (int j = 0; j < 8; ++j) acc[jt][j] = 0.f;
#pragma unroll
  for (int j = 0; j < 8; ++j) rsum[j] = 0.f;

  const int NKT = Mn / 32;                     // 64 key steps
  stage(0, 0);
  wait_async_lds();
  __syncthreads();

  for (int kt = 0; kt < NKT; ++kt) {
    int buf = kt & 1;
    if (kt + 1 < NKT) stage(kt + 1, buf ^ 1);

#pragma unroll
    for (int sub = 0; sub < 2; ++sub) {
      v16h bk = lds_b_frag(&Kbuf[buf][0], sub);
      v8f z;
#pragma unroll
      for (int j = 0; j < 8; ++j) z[j] = 0.f;
      v8f s = WMMA_F16(aq, bk, z);
      float mv = (float)mrow[kt * 32 + sub * 16 + (int)colL];
#pragma unroll
      for (int j = 0; j < 8; ++j) {
        float sv = fmaxf(s[j] * scale, 0.f);   // relu before softmax
        float p  = mv * __expf(sv);            // fold post-softmax mask
        rsum[j] += p;
        Pb[wave][(j + rAdd) * 32 + sub * 16 + colL] = (half_t)p;
      }
    }
    // Re-fetch 16x32 P tile in A-operand layout (C/D -> A shuffle via LDS).
    v16h ap;
    {
      unsigned klo = (lane < 16u) ? 0u : 8u;
      const half_t* pp = &Pb[wave][colL * 32 + klo];
      v8h lo = *(const v8h*)(pp);
      v8h hi = *(const v8h*)(pp + 16);
#pragma unroll
      for (int i = 0; i < 8; ++i) { ap[i] = lo[i]; ap[8 + i] = hi[i]; }
    }
#pragma unroll
    for (int jt = 0; jt < 8; ++jt) {           // DV = 128 -> 8 column tiles
      v16h bv = lds_b_frag(&Vbuf[buf][0], jt);
      acc[jt] = WMMA_F16(ap, bv, acc[jt]);
    }

    wait_async_lds();                          // next K/V tile resident
    __syncthreads();                           // everyone done with buf
  }

  // Row sums: reduce across the 16 lanes sharing a row group.
#pragma unroll
  for (int j = 0; j < 8; ++j) {
    float v = rsum[j];
    v += __shfl_xor(v, 1, 32);
    v += __shfl_xor(v, 2, 32);
    v += __shfl_xor(v, 4, 32);
    v += __shfl_xor(v, 8, 32);
    rsum[j] = v;
  }

  float g = gamma[0];
#pragma unroll
  for (int jt = 0; jt < 8; ++jt) {
    int c = h * DVn + jt * 16 + (int)colL;
#pragma unroll
    for (int j = 0; j < 8; ++j) {
      int m = qt * 16 + j + (int)rAdd;
      size_t idx = ((size_t)b * Mn + m) * Cn + c;
      out[idx] = g * (acc[jt][j] / (rsum[j] + 1e-12f)) + x[idx];
    }
  }
}

// ---------------------------------------------------------------------------
// Host launcher
// ---------------------------------------------------------------------------
extern "C" void kernel_launch(void* const* d_in, const int* in_sizes, int n_in,
                              void* d_out, int out_size, void* d_ws, size_t ws_size,
                              hipStream_t stream) {
  const float* x     = (const float*)d_in[0];
  const int*   mask  = (const int*)d_in[1];
  const float* Wq    = (const float*)d_in[2];
  const float* bq    = (const float*)d_in[3];
  const float* Wk    = (const float*)d_in[4];
  const float* bk    = (const float*)d_in[5];
  const float* Wv    = (const float*)d_in[6];
  const float* bv    = (const float*)d_in[7];
  const float* gamma = (const float*)d_in[8];
  float* out = (float*)d_out;

  // Workspace layout (bytes):
  //   Wt      : 1536*1024*2       =  3,145,728
  //   biasAll : 1536*4            =      6,144
  //   Q       : 4*8*2048*32*2     =  4,194,304
  //   K       : 4*8*2048*32*2     =  4,194,304
  //   Vt      : 4*8*128*2048*2    = 16,777,216   (total ~28.3 MB)
  char* ws = (char*)d_ws;
  half_t* Wt      = (half_t*)(ws);
  float*  biasAll = (float*)(ws + 3145728);
  half_t* Q       = (half_t*)(ws + 3151872);
  half_t* K       = (half_t*)(ws + 7346176);
  half_t* Vt      = (half_t*)(ws + 11540480);

  pack_weights<<<(NTOT * Cn + 255) / 256, 256, 0, stream>>>(
      Wq, Wk, Wv, bq, bk, bv, Wt, biasAll);

  // 64 row-blocks * 24 column groups, 8 waves (128 rows x 64 cols) per block
  qkv_gemm<<<1536, 256, 0, stream>>>(x, Wt, biasAll, Q, K, Vt);

  // 32 (b,h) * 16 query-blocks, 8 waves (128 queries) per block
  attn<<<512, 256, 0, stream>>>(Q, K, Vt, mask, x, gamma, out);
}
